// Res_SSGR_7739531067683
// MI455X (gfx1250) — matrix operand order repaired and checked by reading, not compiled, not run on hardware
//
#include <hip/hip_runtime.h>
#include <stdint.h>

// ---------------------------------------------------------------------------
// Res_SSGR fused pipeline for MI455X (gfx1250, wave32, WMMA bf16 16x16x32,
// TDM tensor_load_to_lds for residual-tile staging overlapped with WMMA)
//   B=8, C=256, INTER=128, RED=32, N=H*W=16384
// ---------------------------------------------------------------------------

typedef __attribute__((ext_vector_type(16))) __bf16 v16bf;
typedef __attribute__((ext_vector_type(8)))  float  v8f;
typedef __attribute__((ext_vector_type(4)))  unsigned int v4u;
typedef __attribute__((ext_vector_type(8)))  unsigned int v8u;

#define BATCH 8
#define CCH   256
#define INTER 128
#define RED   32
#define NSP   16384
#define SLOPE 0.1f
#define EPS   1e-5f
#define TN    64           // spatial tile width for K3/K6
#define NCHUNK 16          // split-K chunks for attention logits
#define CHLEN (NSP / NCHUNK)

// ---------------- WMMA fragment helpers (layouts per CDNA5 ISA 7.12.2) -----

// A fragment: 16x32 bf16, row-major source, row stride ld (elements).
__device__ __forceinline__ v16bf load_frag_a(const __bf16* base, int row0,
                                             int k0, int ld, int lane) {
  int m = lane & 15, half = lane >> 4;
  const __bf16* p = base + (size_t)(row0 + m) * ld + k0 + half * 8;
  union { v16bf v; uint4 u[2]; } r;
  r.u[0] = *(const uint4*)(p);
  r.u[1] = *(const uint4*)(p + 16);
  return r.v;
}

// B fragment: 32x16 bf16. Source buffer indexed [col][k], row stride ld.
__device__ __forceinline__ v16bf load_frag_b(const __bf16* base, int col0,
                                             int k0, int ld, int lane) {
  int n = lane & 15, half = lane >> 4;
  const __bf16* p = base + (size_t)(col0 + n) * ld + k0 + half * 8;
  union { v16bf v; uint4 u[2]; } r;
  r.u[0] = *(const uint4*)(p);
  r.u[1] = *(const uint4*)(p + 16);
  return r.v;
}

__device__ __forceinline__ v8f wmma_bf16(v16bf a, v16bf b, v8f c) {
  return __builtin_amdgcn_wmma_f32_16x16x32_bf16(false, a, false, b,
                                                 (short)0, c, false, false);
}

// ---------------- TDM: async DMA of a [256 x TN] f32 tile into LDS ---------
// D# per CDNA5 ISA §8.3/8.4: group0 = {count/lds_addr/global_addr/type},
// group1 = {data_size, tensor dims, tile dims, strides}. 2D -> 2 groups.
__device__ __forceinline__ void tdm_load_xtile(unsigned int lds_off,
                                               const float* gptr) {
  uint64_t ga = (uint64_t)(uintptr_t)gptr;
  v4u g0;
  g0[0] = 1u;                                   // count=1, user mode
  g0[1] = lds_off;                              // LDS byte address
  g0[2] = (unsigned int)ga;                     // global addr [31:0]
  g0[3] = (unsigned int)((ga >> 32) & 0x01FFFFFFu) | (2u << 30);  // addr[56:32] | type=2
  v8u g1;
  g1[0] = 0x00020000u;                          // data_size=2 (4B), no flags
  g1[1] = (16384u & 0xFFFFu) << 16;             // tensor_dim0[15:0]=16384
  g1[2] = (256u << 16) | (16384u >> 16);        // dim0[31:16] | tensor_dim1=256
  g1[3] = ((unsigned int)TN << 16);             // tile_dim0=TN
  g1[4] = 256u;                                 // tile_dim1=256, tile_dim2=0
  g1[5] = 16384u;                               // tensor_dim0_stride[31:0]
  g1[6] = 0u;                                   // stride[47:32] | dim1_stride lo
  g1[7] = 0u;
  asm volatile("tensor_load_to_lds %0, %1" :: "s"(g0), "s"(g1) : "memory");
}

// ---------------- K1: spatial mean per (b,c) -------------------------------
__global__ __launch_bounds__(256) void k1_mean(const float* __restrict__ x,
                                               float* __restrict__ ymean) {
  int bc = blockIdx.x;
  int tid = threadIdx.x;
  const float* xp = x + (size_t)bc * NSP;
  float s = 0.f;
  for (int i = tid; i < NSP; i += 256) {
    __builtin_prefetch(xp + i + 4096, 0, 0);  // global_prefetch_b8
    s += xp[i];
  }
  __shared__ float red[256];
  red[tid] = s;
  __syncthreads();
  for (int off = 128; off > 0; off >>= 1) {
    if (tid < off) red[tid] += red[tid + off];
    __syncthreads();
  }
  if (tid == 0) ymean[bc] = red[0] * (1.0f / (float)NSP);
}

// ---------------- K2: SE MLP -> sigmoid gate y[b,c] ------------------------
__global__ __launch_bounds__(256) void k2_se(const float* __restrict__ ymean,
                                             const float* __restrict__ w1,
                                             const float* __restrict__ w2,
                                             float* __restrict__ yg) {
  __shared__ float hid[RED];
  int tid = threadIdx.x;
  for (int b = 0; b < BATCH; ++b) {
    if (tid < RED) {
      float s = 0.f;
      for (int c = 0; c < CCH; ++c) s += ymean[b * CCH + c] * w1[tid * CCH + c];
      hid[tid] = (s > 0.f) ? s : SLOPE * s;
    }
    __syncthreads();
    float s = 0.f;
    for (int r = 0; r < RED; ++r) s += hid[r] * w2[tid * RED + r];
    yg[b * CCH + tid] = 1.0f / (1.0f + __expf(-s));
    __syncthreads();
  }
}

// ---------------- Kprep: stack + convert Wq/Wk/Wv to bf16 [384][256] -------
__global__ __launch_bounds__(256) void kprep_w(const float* __restrict__ wq,
                                               const float* __restrict__ wk,
                                               const float* __restrict__ wv,
                                               __bf16* __restrict__ Wqkv) {
  int r = blockIdx.x, tid = threadIdx.x;
  const float* src = (r < 128) ? (wq + r * CCH)
                   : (r < 256) ? (wk + (r - 128) * CCH)
                               : (wv + (r - 256) * CCH);
  Wqkv[(size_t)r * CCH + tid] = (__bf16)src[tid];
}

// ---------------- K3: QKV projection (WMMA), writes q, v, kT ---------------
// q,v: [b][i][n] bf16 ; kT: [b][n][i] bf16 (transposed at D-store time)
__global__ __launch_bounds__(256) void k3_qkv(const float* __restrict__ x,
                                              const float* __restrict__ yg,
                                              const __bf16* __restrict__ Wqkv,
                                              const float* __restrict__ bq,
                                              const float* __restrict__ bk,
                                              const float* __restrict__ bv,
                                              __bf16* __restrict__ q,
                                              __bf16* __restrict__ v,
                                              __bf16* __restrict__ kT) {
  int b = blockIdx.y, n0 = blockIdx.x * TN, tid = threadIdx.x;
  __shared__ __bf16 xsT[TN][CCH + 8];   // [n][c], gate applied, bf16

  {  // stage xs^T: thread = channel, coalesced reads along n
    int c = tid;
    float g = yg[b * CCH + c];
    const float* xp = x + (size_t)(b * CCH + c) * NSP + n0;
    for (int j = 0; j < TN; j += 4) {
      float4 f = *(const float4*)(xp + j);
      xsT[j + 0][c] = (__bf16)(f.x * g);
      xsT[j + 1][c] = (__bf16)(f.y * g);
      xsT[j + 2][c] = (__bf16)(f.z * g);
      xsT[j + 3][c] = (__bf16)(f.w * g);
    }
  }
  __syncthreads();

  int wave = tid >> 5, lane = tid & 31, nl = lane & 15, half = lane >> 4;
  v8f acc[3][4];
  for (int t = 0; t < 3; ++t)
    for (int nb = 0; nb < 4; ++nb) acc[t][nb] = (v8f)0.f;

  for (int k0 = 0; k0 < CCH; k0 += 32) {
    v16bf aF[3];
    for (int t = 0; t < 3; ++t)
      aF[t] = load_frag_a(Wqkv, (wave + 8 * t) * 16, k0, CCH, lane);
    for (int nb = 0; nb < 4; ++nb) {
      v16bf bF = load_frag_b(&xsT[0][0], nb * 16, k0, CCH + 8, lane);
      for (int t = 0; t < 3; ++t) acc[t][nb] = wmma_bf16(aF[t], bF, acc[t][nb]);
    }
  }

  for (int t = 0; t < 3; ++t) {
    const float* bias = (t == 0) ? bq : (t == 1) ? bk : bv;
    for (int nb = 0; nb < 4; ++nb) {
      int n = n0 + nb * 16 + nl;
      if (t == 1) {  // k -> transposed store: lane holds 8 consecutive i
        union { uint4 u; __bf16 h[8]; } pk;
        for (int g = 0; g < 8; ++g) {
          int ii = wave * 16 + 8 * half + g;
          pk.h[g] = (__bf16)(acc[t][nb][g] + bias[ii]);
        }
        *(uint4*)(kT + ((size_t)(b * NSP + n)) * INTER + wave * 16 + 8 * half) = pk.u;
      } else {
        __bf16* dst = (t == 0) ? q : v;
        for (int g = 0; g < 8; ++g) {
          int ii = wave * 16 + g + 8 * half;
          dst[((size_t)(b * INTER + ii)) * NSP + n] =
              (__bf16)(acc[t][nb][g] + bias[ii]);
        }
      }
    }
  }
}

// ---------------- K4a: split-K partial logits part[ch][b][i][j] ------------
__global__ __launch_bounds__(256) void k4a_logits(const __bf16* __restrict__ q,
                                                  const __bf16* __restrict__ v,
                                                  float* __restrict__ part) {
  int ch = blockIdx.x, b = blockIdx.y, tid = threadIdx.x;
  int wave = tid >> 5, lane = tid & 31, nl = lane & 15, half = lane >> 4;
  const __bf16* qb = q + (size_t)b * INTER * NSP;
  const __bf16* vb = v + (size_t)b * INTER * NSP;

  v8f acc[8];
  for (int jb = 0; jb < 8; ++jb) acc[jb] = (v8f)0.f;

  int nbeg = ch * CHLEN;
  for (int n0 = nbeg; n0 < nbeg + CHLEN; n0 += 32) {
    v16bf aF = load_frag_a(qb, wave * 16, n0, NSP, lane);
    for (int jb = 0; jb < 8; ++jb) {
      v16bf bF = load_frag_b(vb, jb * 16, n0, NSP, lane);  // contraction = n
      acc[jb] = wmma_bf16(aF, bF, acc[jb]);
    }
  }

  float* pp = part + (((size_t)ch * BATCH + b) * INTER) * INTER;
  for (int jb = 0; jb < 8; ++jb)
    for (int g = 0; g < 8; ++g)
      pp[(wave * 16 + g + 8 * half) * INTER + jb * 16 + nl] = acc[jb][g];
}

// ---------------- K4b: reduce partials + row softmax -> att[b][i][j] -------
__global__ __launch_bounds__(128) void k4b_softmax(const float* __restrict__ part,
                                                   float* __restrict__ att) {
  int b = blockIdx.x, i = threadIdx.x;
  float* ar = att + ((size_t)b * INTER + i) * INTER;
  float m = -1e30f;
  for (int j = 0; j < INTER; ++j) {
    float s = 0.f;
    for (int p = 0; p < NCHUNK; ++p)
      s += part[(((size_t)p * BATCH + b) * INTER + i) * INTER + j];
    ar[j] = s;
    m = fmaxf(m, s);
  }
  float ssum = 0.f;
  for (int j = 0; j < INTER; ++j) {
    float e = __expf(ar[j] - m);
    ar[j] = e;
    ssum += e;
  }
  float inv = 1.0f / ssum;
  for (int j = 0; j < INTER; ++j) ar[j] *= inv;
}

// ---------------- K5a: fold BNs -> W2[c][o], bias2[c] ----------------------
__global__ __launch_bounds__(128) void k5a_fold(
    const float* __restrict__ w_out, const float* __restrict__ b_out,
    const float* __restrict__ g_wg, const float* __restrict__ be_wg,
    const float* __restrict__ rm_wg, const float* __restrict__ rv_wg,
    const float* __restrict__ g_out, const float* __restrict__ be_out,
    const float* __restrict__ rm_out, const float* __restrict__ rv_out,
    float* __restrict__ W2, float* __restrict__ bias2) {
  int c = blockIdx.x, o = threadIdx.x;
  float sw = g_wg[o] * rsqrtf(rv_wg[o] + EPS);
  float tw = be_wg[o] - rm_wg[o] * sw;
  float s2 = g_out[c] * rsqrtf(rv_out[c] + EPS);
  float t2 = be_out[c] - rm_out[c] * s2;
  float w = w_out[c * INTER + o];
  W2[c * INTER + o] = s2 * w * sw;
  __shared__ float red[INTER];
  red[o] = w * tw;
  __syncthreads();
  for (int off = 64; off > 0; off >>= 1) {
    if (o < off) red[o] += red[o + off];
    __syncthreads();
  }
  if (o == 0) bias2[c] = s2 * (red[0] + b_out[c]) + t2;
}

// ---------------- K5b: W3 = W2 @ w_wg --------------------------------------
__global__ __launch_bounds__(128) void k5b_w3(const float* __restrict__ W2,
                                              const float* __restrict__ w_wg,
                                              float* __restrict__ W3) {
  int c = blockIdx.x, i = threadIdx.x;
  float s = 0.f;
  for (int o = 0; o < INTER; ++o) s += W2[c * INTER + o] * w_wg[o * INTER + i];
  W3[c * INTER + i] = s;
}

// ---------------- K5c: M3[b][c][j] = sum_i W3[c][i] * att[b][j][i] ---------
__global__ __launch_bounds__(128) void k5c_m3(const float* __restrict__ W3,
                                              const float* __restrict__ att,
                                              __bf16* __restrict__ M3) {
  int c = blockIdx.x, b = blockIdx.y, j = threadIdx.x;
  const float* ar = att + ((size_t)b * INTER + j) * INTER;
  const float* wr = W3 + (size_t)c * INTER;
  float s = 0.f;
  for (int i = 0; i < INTER; ++i) s += wr[i] * ar[i];
  M3[((size_t)b * CCH + c) * INTER + j] = (__bf16)s;
}

// ---------------- K6: out = leaky(M3[b] @ K + bias2 + x*y) -----------------
// Residual x-tile is DMA'd into LDS by the Tensor Data Mover, fully
// overlapped with the WMMA loop; wave0 waits on TENSORcnt before epilogue.
__global__ __launch_bounds__(256) void k6_out(const float* __restrict__ x,
                                              const float* __restrict__ yg,
                                              const __bf16* __restrict__ M3,
                                              const __bf16* __restrict__ kT,
                                              const float* __restrict__ bias2,
                                              float* __restrict__ out) {
  int b = blockIdx.y, n0 = blockIdx.x * TN, tid = threadIdx.x;
  __shared__ float xraw[CCH][TN];  // 64 KB, filled by TDM

  if (tid < 32) {  // one TDM issue per block (wave 0)
    tdm_load_xtile((unsigned int)(uintptr_t)(&xraw[0][0]),
                   x + (size_t)b * CCH * NSP + n0);
  }

  int wave = tid >> 5, lane = tid & 31, nl = lane & 15, half = lane >> 4;
  const __bf16* m3b = M3 + (size_t)b * CCH * INTER;
  const __bf16* ktb = kT + (size_t)b * NSP * INTER;

  v8f acc[2][4];
  for (int h = 0; h < 2; ++h)
    for (int nb = 0; nb < 4; ++nb) acc[h][nb] = (v8f)0.f;

  for (int j0 = 0; j0 < INTER; j0 += 32) {
    v16bf aF0 = load_frag_a(m3b, wave * 16, j0, INTER, lane);
    v16bf aF1 = load_frag_a(m3b, (wave + 8) * 16, j0, INTER, lane);
    for (int nb = 0; nb < 4; ++nb) {
      v16bf bF = load_frag_b(ktb, n0 + nb * 16, j0, INTER, lane);
      acc[0][nb] = wmma_bf16(aF0, bF, acc[0][nb]);
      acc[1][nb] = wmma_bf16(aF1, bF, acc[1][nb]);
    }
  }

  if (tid < 32) __builtin_amdgcn_s_wait_tensorcnt(0);
  __syncthreads();  // TDM data now visible to all waves

  for (int h = 0; h < 2; ++h)
    for (int nb = 0; nb < 4; ++nb)
      for (int g = 0; g < 8; ++g) {
        int c = (wave + 8 * h) * 16 + g + 8 * half;
        int nloc = nb * 16 + nl;
        float xsv = xraw[c][nloc] * yg[b * CCH + c];
        float val = acc[h][nb][g] + bias2[c] + xsv;
        xraw[c][nloc] = (val > 0.f) ? val : SLOPE * val;
      }
  __syncthreads();

  {  // coalesced store
    int c = tid;
    float* op = out + (size_t)(b * CCH + c) * NSP + n0;
    for (int j = 0; j < TN; j += 4) *(float4*)(op + j) = *(float4*)&xraw[c][j];
  }
}

// ---------------------------------------------------------------------------
extern "C" void kernel_launch(void* const* d_in, const int* in_sizes, int n_in,
                              void* d_out, int out_size, void* d_ws,
                              size_t ws_size, hipStream_t stream) {
  (void)in_sizes; (void)n_in; (void)out_size; (void)ws_size;
  const float* x     = (const float*)d_in[0];
  const float* se_w1 = (const float*)d_in[1];
  const float* se_w2 = (const float*)d_in[2];
  const float* wq    = (const float*)d_in[3];
  const float* bq    = (const float*)d_in[4];
  const float* wk    = (const float*)d_in[5];
  const float* bk    = (const float*)d_in[6];
  const float* wv    = (const float*)d_in[7];
  const float* bv    = (const float*)d_in[8];
  const float* w_wg  = (const float*)d_in[9];
  const float* g_wg  = (const float*)d_in[10];
  const float* be_wg = (const float*)d_in[11];
  const float* rm_wg = (const float*)d_in[12];
  const float* rv_wg = (const float*)d_in[13];
  const float* w_out = (const float*)d_in[14];
  const float* b_out = (const float*)d_in[15];
  const float* g_out = (const float*)d_in[16];
  const float* be_out = (const float*)d_in[17];
  const float* rm_out = (const float*)d_in[18];
  const float* rv_out = (const float*)d_in[19];
  float* out = (float*)d_out;

  char* ws = (char*)d_ws;
  size_t off = 0;
  auto take = [&](size_t bytes) { char* p = ws + off; off = (off + bytes + 255) & ~(size_t)255; return p; };
  float*  ymean = (float*)take((size_t)BATCH * CCH * 4);
  float*  yg    = (float*)take((size_t)BATCH * CCH * 4);
  __bf16* Wqkv  = (__bf16*)take((size_t)384 * CCH * 2);
  float*  W2    = (float*)take((size_t)CCH * INTER * 4);
  float*  W3    = (float*)take((size_t)CCH * INTER * 4);
  float*  bias2 = (float*)take((size_t)CCH * 4);
  float*  att   = (float*)take((size_t)BATCH * INTER * INTER * 4);
  float*  part  = (float*)take((size_t)NCHUNK * BATCH * INTER * INTER * 4);
  __bf16* M3    = (__bf16*)take((size_t)BATCH * CCH * INTER * 2);
  __bf16* qb    = (__bf16*)take((size_t)BATCH * INTER * NSP * 2);
  __bf16* vb    = (__bf16*)take((size_t)BATCH * INTER * NSP * 2);
  __bf16* kTb   = (__bf16*)take((size_t)BATCH * NSP * INTER * 2);

  k1_mean<<<BATCH * CCH, 256, 0, stream>>>(x, ymean);
  k2_se<<<1, 256, 0, stream>>>(ymean, se_w1, se_w2, yg);
  kprep_w<<<384, 256, 0, stream>>>(wq, wk, wv, Wqkv);
  k3_qkv<<<dim3(NSP / TN, BATCH), 256, 0, stream>>>(x, yg, Wqkv, bq, bk, bv,
                                                    qb, vb, kTb);
  k4a_logits<<<dim3(NCHUNK, BATCH), 256, 0, stream>>>(qb, vb, part);
  k4b_softmax<<<BATCH, 128, 0, stream>>>(part, att);
  k5a_fold<<<CCH, 128, 0, stream>>>(w_out, b_out, g_wg, be_wg, rm_wg, rv_wg,
                                    g_out, be_out, rm_out, rv_out, W2, bias2);
  k5b_w3<<<CCH, 128, 0, stream>>>(W2, w_wg, W3);
  k5c_m3<<<dim3(CCH, BATCH), 128, 0, stream>>>(W3, att, M3);
  k6_out<<<dim3(NSP / TN, BATCH), 256, 0, stream>>>(x, yg, M3, kTb, bias2, out);
}
